// SPINN_74715251081736
// MI455X (gfx1250) — matrix operand order repaired and checked by reading, not compiled, not run on hardware
//
#include <hip/hip_runtime.h>
#include <hip/hip_bf16.h>
#include <math.h>
#include <stdint.h>

// ---------------------------------------------------------------------------
// SPINN on gfx1250: WMMA f16->f32 GEMMs (2x2 register-blocked, async global->
// LDS A-tile staging) + fused elementwise/stack kernels.
// ---------------------------------------------------------------------------

#define BATCH 512
#define NTOK  128
#define TLEN  255          // 2*N - 1
#define SZ    300
#define TS    64
#define KG    1024         // padded K for g-GEMM: 320+320+320+64
#define NG    256          // 4*TS
#define KR    704          // padded K for r-GEMM: 320+320+64
#define NR    1536         // 1500 padded to /64 (macro-tile N)
#define PAD   320          // 300 padded to /32

typedef __attribute__((ext_vector_type(16))) _Float16 v16h;
typedef __attribute__((ext_vector_type(8)))  float    v8f;

union FragU { v16h v; float4 q[2]; };

__device__ __forceinline__ float sigf(float x) { return 1.0f / (1.0f + expf(-x)); }

// ---- CDNA5 async global->LDS copy (ASYNCcnt-tracked, no VGPR bounce) ------
__device__ __forceinline__ void async_copy_b128(void* lds_ptr, const void* gptr) {
  unsigned lds_addr = (unsigned)(uintptr_t)lds_ptr;   // LDS offset = low 32 bits
  asm volatile("global_load_async_to_lds_b128 %0, %1, off"
               :: "v"(lds_addr), "v"(gptr) : "memory");
}
__device__ __forceinline__ void wait_async0() {
#if __has_builtin(__builtin_amdgcn_s_wait_asynccnt)
  __builtin_amdgcn_s_wait_asynccnt(0);
#else
  asm volatile("s_wait_asynccnt 0x0" ::: "memory");
#endif
}

// ----------------------------- zero fills ----------------------------------
__global__ void k_zero_f32(float* __restrict__ p, size_t n) {
  size_t i = (size_t)blockIdx.x * blockDim.x + threadIdx.x;
  if (i < n) p[i] = 0.0f;
}
__global__ void k_zero_f16(_Float16* __restrict__ p, size_t n) {
  size_t i = (size_t)blockIdx.x * blockDim.x + threadIdx.x;
  if (i < n) p[i] = (_Float16)0.0f;
}

// ------------------- weight prep: Wg = [Wb@Wih | Ws1@Wih | Ws2@Wih | Whh] --
__global__ __launch_bounds__(256) void k_prep_wg(
    const float* __restrict__ Wb, const float* __restrict__ Ws1,
    const float* __restrict__ Ws2, const float* __restrict__ Wih,
    const float* __restrict__ Whh, _Float16* __restrict__ Wg) {
  int idx = blockIdx.x * 256 + threadIdx.x;       // KG*NG threads
  if (idx >= KG * NG) return;
  int row = idx >> 8;          // 0..1023
  int col = idx & 255;         // 0..255
  float v = 0.0f;
  const float* src = nullptr;
  int r = -1;
  if (row < 300)                       { src = Wb;  r = row; }
  else if (row >= 320 && row < 620)    { src = Ws1; r = row - 320; }
  else if (row >= 640 && row < 940)    { src = Ws2; r = row - 640; }
  else if (row >= 960)                 { v = Whh[(row - 960) * NG + col]; }
  if (src) {
    float acc = 0.0f;
    for (int k = 0; k < NG; ++k)
      acc = fmaf(src[r * NG + k], Wih[k * NG + col], acc);
    v = acc;
  }
  Wg[row * NG + col] = (_Float16)v;
}

// ------------------- weight prep: Wr = [WL | WR | Wtr] (f16, padded) -------
__global__ __launch_bounds__(256) void k_prep_wr(
    const float* __restrict__ WL, const float* __restrict__ WR,
    const float* __restrict__ Wtr, _Float16* __restrict__ Wr) {
  int idx = blockIdx.x * 256 + threadIdx.x;       // KR*NR threads
  if (idx >= KR * NR) return;
  int row = idx / NR;
  int col = idx % NR;
  float v = 0.0f;
  if (col < 1500) {
    if (row < 300)                    v = WL[row * 1500 + col];
    else if (row >= 320 && row < 620) v = WR[(row - 320) * 1500 + col];
    else if (row >= 640 && row < 704) v = Wtr[(row - 640) * 1500 + col];
  }
  Wr[row * NR + col] = (_Float16)v;
}

// ------------------- initial gather: sptr=bptr=0, buf_top = tokens[:,0] ----
__global__ void k_init(const float* __restrict__ tokens,
                       _Float16* __restrict__ Ag, float* __restrict__ buftop,
                       int* __restrict__ sptr, int* __restrict__ bptr) {
  int b = blockIdx.x;
  const float* tok0 = tokens + (size_t)b * NTOK * (2 * SZ);
  for (int j = threadIdx.x; j < SZ; j += blockDim.x)
    Ag[(size_t)b * KG + j] = (_Float16)tok0[j];
  for (int e = threadIdx.x; e < 2 * SZ; e += blockDim.x)
    buftop[(size_t)b * (2 * SZ) + e] = tok0[e];
  if (threadIdx.x == 0) { sptr[b] = 0; bptr[b] = 0; }
}

// ------------------- WMMA GEMM: C[M,N] = A[M,K] * B[K,N]  (f16 in, f32 out)
// 8 waves / WG as 4x2; macro-tile 128x64; each wave computes a 32x32 block
// (2x2 WMMA tiles) so A/B fragments are reused across tiles. A tile staged
// via async global->LDS copies; B tile loaded + transposed into LDS.
__global__ __launch_bounds__(256) void k_gemm_f16(
    const _Float16* __restrict__ A, const _Float16* __restrict__ B,
    float* __restrict__ C, int M, int N, int K) {
  __shared__ __align__(16) _Float16 As[128 * 40];  // [row][k], stride 40 halves
  __shared__ __align__(16) _Float16 Bs[64 * 40];   // transposed: [n][k]
  const int tid  = threadIdx.x;
  const int lane = tid & 31;
  const int wave = tid >> 5;
  const int m0 = blockIdx.y * 128;
  const int n0 = blockIdx.x * 64;
  const int wrow = (wave >> 1) * 32;               // 0,32,64,96
  const int wcol = (wave & 1) * 32;                // 0,32

  v8f acc[2][2];
#pragma unroll
  for (int i = 0; i < 2; ++i)
#pragma unroll
    for (int j = 0; j < 2; ++j)
      acc[i][j] = (v8f){0.f, 0.f, 0.f, 0.f, 0.f, 0.f, 0.f, 0.f};

  const int a_row = tid >> 1;                      // 0..127
  const int a_col = (tid & 1) * 16;                // 0,16
  const int b_k   = tid >> 3;                      // 0..31
  const int b_n   = (tid & 7) * 8;                 // 0..56

  for (int kb = 0; kb < K; kb += 32) {
    // A tile: 128x32 halves, async global->LDS (16B x2 per thread)
    const _Float16* ap = A + (size_t)(m0 + a_row) * K + kb + a_col;
    async_copy_b128(&As[a_row * 40 + a_col], ap);
    async_copy_b128(&As[a_row * 40 + a_col + 8], ap + 8);
    // B tile: 32x64 halves, load (16B/thread) + transpose into Bs[n][k]
    const _Float16* bp = B + (size_t)(kb + b_k) * N + n0 + b_n;
    float4 braw = *reinterpret_cast<const float4*>(bp);
    const _Float16* bh = reinterpret_cast<const _Float16*>(&braw);
#pragma unroll
    for (int q = 0; q < 8; ++q) Bs[(b_n + q) * 40 + b_k] = bh[q];
    if (kb + 32 < K) __builtin_prefetch(bp + 32 * N, 0, 1); // global_prefetch_b8
    wait_async0();
    __syncthreads();

    // A frags: lane l<16 -> row l, K {0..7,16..23}; l>=16 -> {8..15,24..31}
    FragU af[2], bf[2];
    const int k2 = (lane >> 4) * 8;
    const int k3 = (lane >> 4) * 16;
#pragma unroll
    for (int ti = 0; ti < 2; ++ti) {
      const int arow = wrow + ti * 16 + (lane & 15);
      af[ti].q[0] = *reinterpret_cast<const float4*>(&As[arow * 40 + k2]);
      af[ti].q[1] = *reinterpret_cast<const float4*>(&As[arow * 40 + 16 + k2]);
      const int bcol = wcol + ti * 16 + (lane & 15);
      bf[ti].q[0] = *reinterpret_cast<const float4*>(&Bs[bcol * 40 + k2]);
      bf[ti].q[1] = *reinterpret_cast<const float4*>(&Bs[bcol * 40 + 16 + k2]);
    }
#pragma unroll
    for (int ti = 0; ti < 2; ++ti)
#pragma unroll
      for (int tj = 0; tj < 2; ++tj)
        acc[ti][tj] = __builtin_amdgcn_wmma_f32_16x16x32_f16(
            false, af[ti].v, false, bf[tj].v, (short)0, acc[ti][tj],
            false, false);
    __syncthreads();
  }
  // C layout per tile: vgpr j -> row +(lane>>4)*8 + j, col +(lane&15)
#pragma unroll
  for (int ti = 0; ti < 2; ++ti) {
    const int crow0 = m0 + wrow + ti * 16 + (lane >> 4) * 8;
#pragma unroll
    for (int tj = 0; tj < 2; ++tj) {
      const int ccol = n0 + wcol + tj * 16 + (lane & 15);
#pragma unroll
      for (int j = 0; j < 8; ++j)
        C[(size_t)(crow0 + j) * N + ccol] = acc[ti][tj][j];
    }
  }
}

// ------------------- tracking LSTM elementwise -----------------------------
__global__ __launch_bounds__(256) void k_lstm(
    const float* __restrict__ g, float* __restrict__ th, float* __restrict__ tc,
    _Float16* __restrict__ Ag, _Float16* __restrict__ Ar) {
  int idx = blockIdx.x * 256 + threadIdx.x;       // BATCH*TS
  if (idx >= BATCH * TS) return;
  int b = idx >> 6, j = idx & 63;
  const float* gr = g + (size_t)b * NG;
  float gi = gr[j], gf = gr[64 + j], gg = gr[128 + j], go = gr[192 + j];
  float tcn = sigf(gf) * tc[idx] + sigf(gi) * tanhf(gg);
  float thn = sigf(go) * tanhf(tcn);
  tc[idx] = tcn;
  th[idx] = thn;
  Ar[(size_t)b * KR + 640 + j] = (_Float16)thn;   // th_new -> r-GEMM (this step)
  Ag[(size_t)b * KG + 960 + j] = (_Float16)thn;   // th_new -> g-GEMM (next step)
}

// ------------------- logits + NLL accumulation -----------------------------
__global__ void k_logits(const float* __restrict__ th, const float* __restrict__ Wt,
                         const float* __restrict__ bt, const int* __restrict__ trans,
                         int t, float* __restrict__ nll) {
  int b = blockIdx.x * blockDim.x + threadIdx.x;
  if (b >= BATCH) return;
  float s0 = bt[0], s1 = bt[1];
  const float* h = th + (size_t)b * TS;
  for (int j = 0; j < TS; ++j) {
    s0 = fmaf(h[j], Wt[j * 2 + 0], s0);
    s1 = fmaf(h[j], Wt[j * 2 + 1], s1);
  }
  int label = trans[(size_t)b * TLEN + t];
  float m = fmaxf(s0, s1);
  float lse = m + logf(expf(s0 - m) + expf(s1 - m));
  float lp = (label == 0 ? s0 : s1) - lse;
  atomicAdd(nll, -lp);
}

// ------------------- TreeLSTM compose + stack update + next gather ---------
__global__ __launch_bounds__(128) void k_compose(
    const float* __restrict__ r, const float* __restrict__ bL,
    float* __restrict__ s1c, float* __restrict__ s2c,
    float* __restrict__ buftop, float* __restrict__ stack,
    int* __restrict__ sptr, int* __restrict__ bptr,
    const int* __restrict__ trans, int t, const float* __restrict__ tokens,
    _Float16* __restrict__ Ag, _Float16* __restrict__ Ar) {
  __shared__ float red[2 * SZ];
  __shared__ int s_nsp, s_nbp;
  const int b = blockIdx.x;
  const int tid = threadIdx.x;
  const int sp = sptr[b];
  const int is_shift = (trans[(size_t)b * TLEN + t] == 0);

  // phase 1: compose reduced item
  const float* rr = r + (size_t)b * NR;
  for (int j = tid; j < SZ; j += 128) {
    float ra  = rr[j]            + bL[j];
    float ri  = rr[300 + j]      + bL[300 + j];
    float rf1 = rr[600 + j]      + bL[600 + j];
    float rf2 = rr[900 + j]      + bL[900 + j];
    float ro  = rr[1200 + j]     + bL[1200 + j];
    float rc = tanhf(ra) * sigf(ri) + sigf(rf1) * s2c[(size_t)b * SZ + j]
                                    + sigf(rf2) * s1c[(size_t)b * SZ + j];
    float rh = sigf(ro) * tanhf(rc);
    red[j] = rh;
    red[SZ + j] = rc;
  }
  __syncthreads();

  // phase 2: write stack row, update pointers
  int widx = is_shift ? (sp < NTOK - 1 ? sp : NTOK - 1)
                      : (sp - 2 < 0 ? 0 : (sp - 2 < NTOK - 1 ? sp - 2 : NTOK - 1));
  float* srow = stack + ((size_t)b * NTOK + widx) * (2 * SZ);
  const float* bt_row = buftop + (size_t)b * (2 * SZ);
  for (int e = tid; e < 2 * SZ; e += 128)
    srow[e] = is_shift ? bt_row[e] : red[e];
  if (tid == 0) {
    int nsp = is_shift ? sp + 1 : sp - 1;
    int nbp = bptr[b] + is_shift;
    sptr[b] = nsp; bptr[b] = nbp;
    s_nsp = nsp;   s_nbp = nbp;
  }
  __threadfence_block();
  __syncthreads();

  // phase 3: gather inputs for the next step
  const int nsp = s_nsp;
  const int btr = (s_nbp < NTOK - 1) ? s_nbp : NTOK - 1;
  const float* tok = tokens + ((size_t)b * NTOK + btr) * (2 * SZ);
  const float* r1 = stack + ((size_t)b * NTOK + (nsp >= 1 ? nsp - 1 : 0)) * (2 * SZ);
  const float* r2 = stack + ((size_t)b * NTOK + (nsp >= 2 ? nsp - 2 : 0)) * (2 * SZ);
  for (int j = tid; j < SZ; j += 128) {
    float bh  = tok[j];
    float s1h = (nsp >= 1) ? r1[j] : 0.0f;
    float s2h = (nsp >= 2) ? r2[j] : 0.0f;
    float c1  = (nsp >= 1) ? r1[SZ + j] : 0.0f;
    float c2  = (nsp >= 2) ? r2[SZ + j] : 0.0f;
    Ag[(size_t)b * KG + j]           = (_Float16)bh;
    Ag[(size_t)b * KG + PAD + j]     = (_Float16)s1h;
    Ag[(size_t)b * KG + 2 * PAD + j] = (_Float16)s2h;
    Ar[(size_t)b * KR + j]           = (_Float16)s2h;
    Ar[(size_t)b * KR + PAD + j]     = (_Float16)s1h;
    s1c[(size_t)b * SZ + j] = c1;
    s2c[(size_t)b * SZ + j] = c2;
  }
  for (int e = tid; e < 2 * SZ; e += 128)
    buftop[(size_t)b * (2 * SZ) + e] = tok[e];
}

// ------------------- final output ------------------------------------------
__global__ void k_final(const float* __restrict__ stack, const int* __restrict__ sptr,
                        const float* __restrict__ nll, float* __restrict__ out) {
  size_t i = (size_t)blockIdx.x * blockDim.x + threadIdx.x;
  if (i == 0) out[(size_t)BATCH * 2 * SZ] = nll[0] * (1.0f / (float)(TLEN * BATCH));
  if (i >= (size_t)BATCH * 2 * SZ) return;
  int b = (int)(i / (2 * SZ));
  int e = (int)(i % (2 * SZ));
  int fp = sptr[b] - 1;
  fp = fp < 0 ? 0 : (fp > NTOK - 1 ? NTOK - 1 : fp);
  out[i] = stack[((size_t)b * NTOK + fp) * (2 * SZ) + e];
}

// ---------------------------------------------------------------------------
extern "C" void kernel_launch(void* const* d_in, const int* in_sizes, int n_in,
                              void* d_out, int out_size, void* d_ws, size_t ws_size,
                              hipStream_t stream) {
  const float* tokens = (const float*)d_in[0];
  const float* Wb     = (const float*)d_in[1];
  const float* Ws1    = (const float*)d_in[2];
  const float* Ws2    = (const float*)d_in[3];
  const float* W_ih   = (const float*)d_in[4];
  const float* W_hh   = (const float*)d_in[5];
  const float* Wt     = (const float*)d_in[6];
  const float* bt     = (const float*)d_in[7];
  const float* WL     = (const float*)d_in[8];
  const float* WR     = (const float*)d_in[9];
  const float* Wtr    = (const float*)d_in[10];
  const float* bL     = (const float*)d_in[11];
  const int*   trans  = (const int*)d_in[12];
  float* out = (float*)d_out;

  // workspace carve-up (256B aligned)
  char* base = (char*)d_ws;
  size_t off = 0;
  auto take = [&](size_t bytes) -> char* {
    char* p = base + off;
    off = (off + bytes + 255) & ~(size_t)255;
    return p;
  };
  float*    stack  = (float*)take((size_t)BATCH * NTOK * 2 * SZ * 4);
  float*    g_buf  = (float*)take((size_t)BATCH * NG * 4);
  float*    r_buf  = (float*)take((size_t)BATCH * NR * 4);
  float*    th     = (float*)take((size_t)BATCH * TS * 4);   // th,tc,s1c,s2c,nll
  float*    tc     = (float*)take((size_t)BATCH * TS * 4);   //  zeroed as one block
  float*    s1c    = (float*)take((size_t)BATCH * SZ * 4);
  float*    s2c    = (float*)take((size_t)BATCH * SZ * 4);
  float*    nll    = (float*)take(256);                      // 1 float used
  float*    buftop = (float*)take((size_t)BATCH * 2 * SZ * 4);
  _Float16* Ag     = (_Float16*)take((size_t)BATCH * KG * 2);
  _Float16* Ar     = (_Float16*)take((size_t)BATCH * KR * 2);
  _Float16* Wg     = (_Float16*)take((size_t)KG * NG * 2);
  _Float16* Wr     = (_Float16*)take((size_t)KR * NR * 2);
  int*      sptr   = (int*)take(BATCH * 4);
  int*      bptr   = (int*)take(BATCH * 4);
  (void)ws_size; (void)in_sizes; (void)n_in; (void)out_size;

  // ---- per-call (re)initialization: deterministic, graph-replay safe ------
  {
    size_t nz = (size_t)(((char*)nll + 256) - (char*)th) / 4;
    k_zero_f32<<<(unsigned)((nz + 255) / 256), 256, 0, stream>>>(th, nz);
    size_t nag = (size_t)BATCH * KG, nar = (size_t)BATCH * KR;
    k_zero_f16<<<(unsigned)((nag + 255) / 256), 256, 0, stream>>>(Ag, nag);
    k_zero_f16<<<(unsigned)((nar + 255) / 256), 256, 0, stream>>>(Ar, nar);
    k_init<<<BATCH, 128, 0, stream>>>(tokens, Ag, buftop, sptr, bptr);
    k_prep_wg<<<(KG * NG + 255) / 256, 256, 0, stream>>>(Wb, Ws1, Ws2, W_ih, W_hh, Wg);
    k_prep_wr<<<(KR * NR + 255) / 256, 256, 0, stream>>>(WL, WR, Wtr, Wr);
  }

  const dim3 grid_g(NG / 64, BATCH / 128);   // (4, 4)
  const dim3 grid_r(NR / 64, BATCH / 128);   // (24, 4)

  for (int t = 0; t < TLEN; ++t) {
    k_gemm_f16<<<grid_g, 256, 0, stream>>>(Ag, Wg, g_buf, BATCH, NG, KG);
    k_lstm<<<(BATCH * TS + 255) / 256, 256, 0, stream>>>(g_buf, th, tc, Ag, Ar);
    k_logits<<<(BATCH + 255) / 256, 256, 0, stream>>>(th, Wt, bt, trans, t, nll);
    k_gemm_f16<<<grid_r, 256, 0, stream>>>(Ar, Wr, r_buf, BATCH, NR, KR);
    k_compose<<<BATCH, 128, 0, stream>>>(r_buf, bL, s1c, s2c, buftop, stack,
                                         sptr, bptr, trans, t, tokens, Ag, Ar);
  }
  k_final<<<(BATCH * 2 * SZ + 255) / 256, 256, 0, stream>>>(stack, sptr, nll, out);
}
